// OverfitAudioNetwork_27273042329922
// MI455X (gfx1250) — compile-verified, matrix operand order checked by reading, not run on hardware
//
#include <hip/hip_runtime.h>
#include <hip/hip_bf16.h>
#include <math.h>

// ---------------- problem constants (from reference) ----------------
#define NSAMP   131072      // N_SAMPLES
#define NFR_SIG 256         // N_FRAMES (signal upsample frames)
#define WS      2048        // window size
#define STEP    1024        // hop
#define CCH     32          // channels
#define KBINS   1025        // rfft bins
#define NLAYERS 3
#define BB      4           // batch
#define NF      128         // STFT frames per channel
#define KPAD    1040        // padded N dim for T matrix (65 tiles of 16)
#define KKPAD   1056        // padded K dim (33 steps of 32)
#define FRSTRIDE 2050       // floats per (b,c,f) spectrum region (1025 complex)

typedef __attribute__((ext_vector_type(16))) __bf16 v16bf;
typedef __attribute__((ext_vector_type(8)))  __bf16 v8bf;
typedef __attribute__((ext_vector_type(8)))  float  v8f;

#define TWO_PI 6.28318530717958647692f

// ---------------- us = sparse-impulse * decay-envelope * noise ----------------
// fft_convolve(upsample_with_holes(sig), env) has zero overlap (spacing==len==512)
__global__ __launch_bounds__(256) void k_build_us(const float* __restrict__ sig,
                                                  const float* __restrict__ noise,
                                                  const float* __restrict__ decays,
                                                  float* __restrict__ us) {
    long i = (long)blockIdx.x * 256 + threadIdx.x;
    const long total = (long)BB * CCH * NSAMP;
    if (i >= total) return;
    int  t  = (int)(i % NSAMP);
    long bc = i / NSAMP;
    int  c  = (int)(bc % CCH);
    int  b  = (int)(bc / CCH);
    int  f  = t >> 9;          // /512
    int  s  = t & 511;
    float env = powf(1.0f - (float)s / 511.0f, decays[c]);   // linspace(1,0,512)^decay
    us[i] = sig[((long)b * CCH + c) * NFR_SIG + f] * env * noise[t];
}

__global__ __launch_bounds__(256) void k_zero(float* __restrict__ p, long n) {
    long i = (long)blockIdx.x * 256 + threadIdx.x;
    if (i < n) p[i] = 0.0f;
}

// ---------------- softmax-weighted channel-sum accumulation ----------------
__global__ __launch_bounds__(256) void k_accum(const float* __restrict__ buf,
                                               const float* __restrict__ mixer,
                                               float* __restrict__ acc, int widx) {
    long i = (long)blockIdx.x * 256 + threadIdx.x;
    const long total = (long)BB * NSAMP;
    if (i >= total) return;
    int t = (int)(i % NSAMP);
    int b = (int)(i / NSAMP);
    float m0 = mixer[0], m1 = mixer[1], m2 = mixer[2], m3 = mixer[3];
    float mx = fmaxf(fmaxf(m0, m1), fmaxf(m2, m3));
    float e0 = __expf(m0 - mx), e1 = __expf(m1 - mx), e2 = __expf(m2 - mx), e3 = __expf(m3 - mx);
    float den = e0 + e1 + e2 + e3;
    float wv = (widx == 0 ? e0 : widx == 1 ? e1 : widx == 2 ? e2 : e3) / den;
    float s = 0.0f;
    #pragma unroll
    for (int c = 0; c < CCH; ++c)
        s += buf[((long)b * CCH + c) * NSAMP + t];
    acc[i] += wv * s;
}

// ---------------- channel mixing: y[b,d,t] = sum_c x[b,c,t] * M[c,d] ----------------
__global__ __launch_bounds__(256) void k_mix(const float* __restrict__ x,
                                             const float* __restrict__ M,
                                             float* __restrict__ y) {
    __shared__ float sM[CCH * CCH];
    for (int j = threadIdx.x; j < CCH * CCH; j += 256) sM[j] = M[j];
    __syncthreads();
    long i = (long)blockIdx.x * 256 + threadIdx.x;   // over B*NSAMP
    if (i >= (long)BB * NSAMP) return;
    int t = (int)(i % NSAMP);
    int b = (int)(i / NSAMP);
    float accv[CCH];
    #pragma unroll
    for (int d = 0; d < CCH; ++d) accv[d] = 0.0f;
    const float* xb = x + (long)b * CCH * NSAMP + t;
    #pragma unroll 4
    for (int c = 0; c < CCH; ++c) {
        __builtin_prefetch(xb + (long)(c + 4) * NSAMP, 0, 0);  // global_prefetch_b8
        float xv = xb[(long)c * NSAMP];
        #pragma unroll
        for (int d = 0; d < CCH; ++d) accv[d] += sM[c * CCH + d] * xv;
    }
    float* yb = y + (long)b * CCH * NSAMP + t;
    #pragma unroll
    for (int d = 0; d < CCH; ++d) yb[(long)d * NSAMP] = accv[d];
}

// ---------------- bf16 pre-pack (zero padded) for the WMMA GEMM ----------------
// Abf: (layer, row 0..31, k 0..1055)  row-major.
// Bsw: WMMA-lane-ordered swizzle: Bsw[(kblk*KPAD + col)*8 + r] = B[8*kblk + r][col]
// so a lane's 8 B values for one K-run are one contiguous 16-byte b128 load.
__global__ __launch_bounds__(256) void k_pack_A(const float* __restrict__ transfers,
                                                __bf16* __restrict__ Abf) {
    int i = blockIdx.x * 256 + threadIdx.x;
    if (i >= NLAYERS * CCH * KKPAD) return;
    int k  = i % KKPAD;
    int rc = i / KKPAD;
    float v = (k < KBINS) ? transfers[(long)rc * KBINS + k] : 0.0f;
    Abf[i] = (__bf16)v;
}
__global__ __launch_bounds__(256) void k_pack_B(const float* __restrict__ fb,
                                                __bf16* __restrict__ Bsw) {
    long i = (long)blockIdx.x * 256 + threadIdx.x;
    if (i >= (long)KKPAD * KPAD) return;
    int col = (int)(i % KPAD);
    int k   = (int)(i / KPAD);
    float v = (k < KBINS && col < KBINS) ? fb[(long)k * KBINS + col] : 0.0f;
    int kblk = k >> 3, r = k & 7;
    Bsw[((long)kblk * KPAD + col) * 8 + r] = (__bf16)v;
}

// ---------------- T[l] = transfers[l](32x1025) @ filter_bank(1025x1025), bf16 WMMA ----------------
// One wave per 16x16 output tile; K padded to 1056 (33 unguarded steps of 32, unroll 3).
// A lane layout (16x32 bf16): lane m=l&15, half h=l>>4; elems 0..7 = K 8h..8h+7,
// elems 8..15 = K 16+8h..16+8h+7 (two contiguous runs -> two b128 loads).
// B (32x16): lane n=l&15, same K split -> two b128 loads from the swizzled layout.
// D: m=8h+r, n=l&15.
__global__ __launch_bounds__(32) void k_transfer_wmma(const __bf16* __restrict__ Abf,
                                                      const __bf16* __restrict__ Bsw,
                                                      float* __restrict__ Tm) {
    const int lane  = threadIdx.x;
    const int ntile = blockIdx.x;   // 0..64
    const int mtile = blockIdx.y;   // 0..1
    const int layer = blockIdx.z;   // 0..2
    const int ml = lane & 15;
    const int h  = lane >> 4;
    const int m0 = mtile * 16;
    const int n0 = ntile * 16;
    const __bf16* Ar = Abf + ((long)layer * CCH + (m0 + ml)) * KKPAD;   // 16B-aligned rows
    const int bcol = n0 + ml;
    const __bf16* Br = Bsw + (long)bcol * 8;   // + kblk*KPAD*8 per step
    v8f acc = {0.f, 0.f, 0.f, 0.f, 0.f, 0.f, 0.f, 0.f};

    #pragma unroll 3
    for (int ks = 0; ks < 33; ++ks) {
        const int k0   = ks * 32 + 8 * h;       // first K-run
        const int kb0  = ks * 4 + h;            // k0 >> 3
        v8bf a0 = *(const v8bf*)(Ar + k0);                              // b128
        v8bf a1 = *(const v8bf*)(Ar + k0 + 16);                         // b128
        v8bf b0 = *(const v8bf*)(Br + (long)kb0 * (KPAD * 8));          // b128
        v8bf b1 = *(const v8bf*)(Br + (long)(kb0 + 2) * (KPAD * 8));    // b128
        v16bf av, bv;
        #pragma unroll
        for (int r = 0; r < 8; ++r) {
            av[r] = a0[r];  av[8 + r] = a1[r];
            bv[r] = b0[r];  bv[8 + r] = b1[r];
        }
        acc = __builtin_amdgcn_wmma_f32_16x16x32_bf16(false, av, false, bv,
                                                      (short)0, acc, false, false);
    }
    float* D = Tm + (long)layer * CCH * KPAD;
    #pragma unroll
    for (int r = 0; r < 8; ++r) {
        int mm = 8 * h + r;
        D[(long)(m0 + mm) * KPAD + (n0 + ml)] = acc[r];
    }
}

// ---------------- shared 2048-pt radix-2 DIT FFT (data pre-bit-reversed) ----------------
__device__ __forceinline__ void fft2048(float2* d, int tid, float sgn) {
    #pragma unroll 1
    for (int s = 1; s <= 11; ++s) {
        const int m = 1 << s, half = m >> 1;
        for (int idx = tid; idx < 1024; idx += 256) {
            int grp = idx >> (s - 1);
            int j   = idx & (half - 1);
            int pos = grp * m + j;
            float ang = sgn * TWO_PI * (float)j / (float)m;
            float sw, cw;
            __sincosf(ang, &sw, &cw);
            float2 a = d[pos], b = d[pos + half];
            float2 tb = make_float2(b.x * cw - b.y * sw, b.x * sw + b.y * cw);
            d[pos]        = make_float2(a.x + tb.x, a.y + tb.y);
            d[pos + half] = make_float2(a.x - tb.x, a.y - tb.y);
        }
        __syncthreads();
    }
}

// ---------------- forward STFT: async-stage frame -> hann -> rfft(2048) ----------------
__global__ __launch_bounds__(256) void k_fft_fwd(const float* __restrict__ y,
                                                 float* __restrict__ spec) {
    __shared__ __align__(16) float2 d[WS];
    __shared__ __align__(16) float  stage[WS];
    const int bcf = blockIdx.x;          // (b*C+c)*NF + f
    const int f   = bcf & (NF - 1);
    const int bc  = bcf >> 7;
    const int tid = threadIdx.x;
    const float* src = y + (long)bc * NSAMP;
    const long  base = (long)f * STEP;
    const unsigned lds_base = (unsigned)(uintptr_t)(void*)stage;  // LDS byte offset (low 32b)

    // Stage the frame into LDS via CDNA5 async global->LDS (ASYNCcnt-tracked).
    // Tail region (t >= NSAMP) starts at a multiple of 1024 floats -> b128-exact.
    for (int i = 4 * tid; i < WS; i += 1024) {
        long t = base + i;
        if (t < NSAMP) {
            unsigned long long ga = (unsigned long long)(uintptr_t)(src + t);
            unsigned la = lds_base + (unsigned)i * 4u;
            asm volatile("global_load_async_to_lds_b128 %0, %1, off"
                         :: "v"(la), "v"(ga) : "memory");
        } else {
            stage[i] = 0.f; stage[i + 1] = 0.f; stage[i + 2] = 0.f; stage[i + 3] = 0.f;
        }
    }
    asm volatile("s_wait_asynccnt 0x0" ::: "memory");
    __syncthreads();

    for (int i = tid; i < WS; i += 256) {
        float w = 0.5f - 0.5f * __cosf(TWO_PI * (float)i / (float)WS); // periodic hann
        unsigned r = __brev((unsigned)i) >> 21;                        // 11-bit reversal
        d[r] = make_float2(stage[i] * w, 0.0f);
    }
    __syncthreads();
    fft2048(d, tid, -1.0f);
    float* out = spec + (long)bcf * FRSTRIDE;
    for (int k = tid; k <= 1024; k += 256) {
        out[2 * k]     = d[k].x;
        out[2 * k + 1] = d[k].y;
    }
}

// ---------------- per-bin linear recurrence over frames: out[f] = (s[f]+out[f-1])*T ----------------
__global__ __launch_bounds__(256) void k_recur(float* __restrict__ spec,
                                               const float* __restrict__ Tm) {
    long i = (long)blockIdx.x * 256 + threadIdx.x;   // over B*C*KBINS
    if (i >= (long)BB * CCH * KBINS) return;
    int  k  = (int)(i % KBINS);
    long bc = i / KBINS;
    int  c  = (int)(bc % CCH);
    float tv = Tm[(long)c * KPAD + k];
    float2 a = make_float2(0.f, 0.f);
    float2* p = (float2*)(spec + bc * (long)NF * FRSTRIDE) + k;   // 8B aligned
    #pragma unroll 4
    for (int f = 0; f < NF; ++f) {
        float2 s = *p;
        a.x = (s.x + a.x) * tv;
        a.y = (s.y + a.y) * tv;
        *p = a;
        p += FRSTRIDE / 2;
    }
}

// ---------------- irfft(2048) + synthesis hann, in-place into spec region ----------------
__global__ __launch_bounds__(256) void k_fft_inv(float* __restrict__ spec) {
    __shared__ __align__(16) float2 d[WS];
    __shared__ __align__(16) float  stage[FRSTRIDE];
    const int bcf = blockIdx.x;
    const int tid = threadIdx.x;
    float* fr = spec + (long)bcf * FRSTRIDE;
    const unsigned lds_base = (unsigned)(uintptr_t)(void*)stage;

    // async-stage the 1025 complex bins
    for (int i = tid; i < FRSTRIDE; i += 256) {
        unsigned long long ga = (unsigned long long)(uintptr_t)(fr + i);
        unsigned la = lds_base + (unsigned)i * 4u;
        asm volatile("global_load_async_to_lds_b32 %0, %1, off"
                     :: "v"(la), "v"(ga) : "memory");
    }
    asm volatile("s_wait_asynccnt 0x0" ::: "memory");
    __syncthreads();

    for (int k = tid; k < WS; k += 256) {
        float re, im;
        if (k <= 1024) { re = stage[2 * k];  im =  stage[2 * k + 1]; }
        else { int kk = WS - k; re = stage[2 * kk]; im = -stage[2 * kk + 1]; }  // conj sym
        unsigned r = __brev((unsigned)k) >> 21;
        d[r] = make_float2(re, im);
    }
    __syncthreads();
    fft2048(d, tid, +1.0f);
    const float scale = 1.0f / (float)WS;
    for (int i = tid; i < WS; i += 256) {
        float w = 0.5f - 0.5f * __cosf(TWO_PI * (float)i / (float)WS);  // synthesis hann
        fr[i] = d[i].x * scale * w;
    }
}

// ---------------- overlap-add (gather, hop=1024) + tanh(gain*x) ----------------
__global__ __launch_bounds__(256) void k_ola(const float* __restrict__ spec,
                                             const float* __restrict__ gains,
                                             float* __restrict__ out) {
    long i = (long)blockIdx.x * 256 + threadIdx.x;   // over B*C*NSAMP
    if (i >= (long)BB * CCH * NSAMP) return;
    int  t  = (int)(i % NSAMP);
    long bc = i / NSAMP;
    int  c  = (int)(bc % CCH);
    int f1 = t >> 10;            // frame whose first half covers t (always < 128)
    int w1 = t & 1023;
    const float* base = spec + bc * (long)NF * FRSTRIDE;
    float s = base[(long)f1 * FRSTRIDE + w1];
    if (f1 > 0) s += base[(long)(f1 - 1) * FRSTRIDE + w1 + STEP];
    out[i] = tanhf(s * gains[c]);
}

// ---------------- pack outputs: [result (B,1,T)] ++ [sig (B,C,256)] ----------------
__global__ __launch_bounds__(256) void k_writeout(const float* __restrict__ acc,
                                                  const float* __restrict__ sig,
                                                  float* __restrict__ out) {
    const long n1 = (long)BB * NSAMP;
    const long n2 = (long)BB * CCH * NFR_SIG;
    long i = (long)blockIdx.x * 256 + threadIdx.x;
    if (i < n1)            out[i] = acc[i];
    else if (i < n1 + n2)  out[i] = sig[i - n1];
}

// ------------------------------------------------------------------------------
extern "C" void kernel_launch(void* const* d_in, const int* in_sizes, int n_in,
                              void* d_out, int out_size, void* d_ws, size_t ws_size,
                              hipStream_t stream) {
    const float* sig       = (const float*)d_in[0];  // (4,32,256)
    const float* noise     = (const float*)d_in[1];  // (131072,)
    const float* decays    = (const float*)d_in[2];  // (32,)
    const float* mixer     = (const float*)d_in[3];  // (4,)
    const float* transfers = (const float*)d_in[4];  // (3,32,1025)
    const float* mmats     = (const float*)d_in[5];  // (3,32,32)
    const float* gains     = (const float*)d_in[6];  // (3,32)
    const float* fb        = (const float*)d_in[7];  // (1025,1025)

    // workspace layout (all section sizes are multiples of 256 B; base is 256B aligned)
    float* bufA = (float*)d_ws;                                  // 67 MB: time-domain ping
    float* bufB = bufA + (long)BB * CCH * NSAMP;                 // 67 MB: time-domain pong
    float* spec = bufB + (long)BB * CCH * NSAMP;                 // 134 MB: spectra / frames
    float* Tm   = spec + (long)BB * CCH * NF * FRSTRIDE;         // 3*32*1040 fp32
    float* acc  = Tm   + (long)NLAYERS * CCH * KPAD;             // (B,T) accumulator
    __bf16* Abf = (__bf16*)(acc + (long)BB * NSAMP);             // 3*32*1056 bf16
    __bf16* Bsw = Abf + (long)NLAYERS * CCH * KKPAD;             // 1056*1040 bf16 (swizzled)

    const long nBCT = (long)BB * CCH * NSAMP;
    const long nBT  = (long)BB * NSAMP;
    const int  nFrBlk = BB * CCH * NF;    // 16384 FFT workgroups

    // us (layer-0 "x") via closed-form sparse convolution
    k_build_us<<<(int)((nBCT + 255) / 256), 256, 0, stream>>>(sig, noise, decays, bufA);
    k_zero<<<(int)((nBT + 255) / 256), 256, 0, stream>>>(acc, nBT);

    // bf16 pre-pack (B in WMMA lane order), then T matrices for all layers via WMMA
    k_pack_A<<<(NLAYERS * CCH * KKPAD + 255) / 256, 256, 0, stream>>>(transfers, Abf);
    k_pack_B<<<(int)(((long)KKPAD * KPAD + 255) / 256), 256, 0, stream>>>(fb, Bsw);
    k_transfer_wmma<<<dim3(65, 2, 3), 32, 0, stream>>>(Abf, Bsw, Tm);

    // accumulate softmax-weighted channel sum of layer-0 input
    k_accum<<<(int)((nBT + 255) / 256), 256, 0, stream>>>(bufA, mixer, acc, 0);

    for (int l = 0; l < NLAYERS; ++l) {
        k_mix<<<(int)((nBT + 255) / 256), 256, 0, stream>>>(bufA, mmats + (long)l * CCH * CCH, bufB);
        k_fft_fwd<<<nFrBlk, 256, 0, stream>>>(bufB, spec);
        k_recur<<<(int)(((long)BB * CCH * KBINS + 255) / 256), 256, 0, stream>>>(
            spec, Tm + (long)l * CCH * KPAD);
        k_fft_inv<<<nFrBlk, 256, 0, stream>>>(spec);
        k_ola<<<(int)((nBCT + 255) / 256), 256, 0, stream>>>(spec, gains + (long)l * CCH, bufA);
        k_accum<<<(int)((nBT + 255) / 256), 256, 0, stream>>>(bufA, mixer, acc, l + 1);
    }

    const long nOut = nBT + (long)BB * CCH * NFR_SIG;
    k_writeout<<<(int)((nOut + 255) / 256), 256, 0, stream>>>(acc, sig, (float*)d_out);
}
